// Self_Attention_45105746542715
// MI455X (gfx1250) — compile-verified, matrix operand order checked by reading, not run on hardware
//
#include <hip/hip_runtime.h>
#include <hip/hip_bf16.h>
#include <math.h>

typedef __attribute__((ext_vector_type(16))) _Float16 v16h;
typedef __attribute__((ext_vector_type(8)))  _Float16 v8h;
typedef __attribute__((ext_vector_type(8)))  float    v8f;

#define DIM    192
#define HEADS  6
#define CH     32          // DIM / HEADS
#define HW     16384       // 128*128
#define BATCH  8
#define BH     (BATCH*HEADS)   // 48
#define NSLICE 32          // K-slices for attn GEMM (occupancy)

// -------- CDNA5 16-bit A/B fragment loader (16 f16 per lane) --------
// Layout (ISA 7.12.2, 16-bit A 16x32): lane L holds row M = L%16;
//   half = L/16: VGPR0-3 = K(half*8 + 0..7), VGPR4-7 = K(16 + half*8 + 0..7).
// B (columns of B = rows of B^T) mirrors this with lane -> N.
__device__ inline v16h load_frag16(const _Float16* __restrict__ row, int kbase, int lane) {
  const int half = (lane >> 4) & 1;
  const v8h a = *reinterpret_cast<const v8h*>(row + kbase + half * 8);
  const v8h b = *reinterpret_cast<const v8h*>(row + kbase + 16 + half * 8);
  v16h r;
  #pragma unroll
  for (int i = 0; i < 8; ++i) { r[i] = a[i]; r[i + 8] = b[i]; }
  return r;
}

// -------- Kernel 0: init accumulators, convert wp -> f16 --------
__global__ void k_init(float* __restrict__ attn, float* __restrict__ ssq,
                       float* __restrict__ ssk, const float* __restrict__ wp,
                       _Float16* __restrict__ wp16) {
  const int i = blockIdx.x * 256 + threadIdx.x;
  if (i < BH * CH * CH) attn[i] = 0.f;
  if (i < BATCH * DIM) { ssq[i] = 0.f; ssk[i] = 0.f; }
  if (i < DIM * DIM) wp16[i] = (_Float16)wp[i];
}

// -------- Kernel 1: fused 3x3 depthwise conv for q,k,v + sum-of-squares --------
// grid (HW/256, DIM, BATCH), block 256. q,k stored [b][c][n] f16; v stored
// channel-minor vT[b][n][192] f16 so final-GEMM B-fragments are contiguous.
// x is read once -> non-temporal loads keep the f16 intermediates L2-resident
// (151 MB of intermediates vs 192 MB global L2).
__global__ void k_dwconv_qkv(const float* __restrict__ x,
                             const float* __restrict__ wq, const float* __restrict__ bq,
                             const float* __restrict__ wk, const float* __restrict__ bk,
                             const float* __restrict__ wv, const float* __restrict__ bv,
                             _Float16* __restrict__ qh, _Float16* __restrict__ kh,
                             _Float16* __restrict__ vT,
                             float* __restrict__ ssq, float* __restrict__ ssk) {
  const int tid = threadIdx.x;
  const int p   = blockIdx.x * 256 + tid;   // pixel 0..16383
  const int c   = blockIdx.y;               // channel
  const int b   = blockIdx.z;               // batch
  const int y = p >> 7, xc = p & 127;

  const float* xb = x + (size_t)(b * DIM + c) * HW;
  float xv[9];
  #pragma unroll
  for (int dy = -1; dy <= 1; ++dy)
    #pragma unroll
    for (int dx = -1; dx <= 1; ++dx) {
      const int yy = y + dy, xx = xc + dx;
      float v = 0.f;
      if (yy >= 0 && yy < 128 && xx >= 0 && xx < 128)
        v = __builtin_nontemporal_load(xb + yy * 128 + xx);
      xv[(dy + 1) * 3 + (dx + 1)] = v;
    }
  float qf = bq[c], kf = bk[c], vf = bv[c];
  #pragma unroll
  for (int i = 0; i < 9; ++i) {
    qf += wq[c * 9 + i] * xv[i];
    kf += wk[c * 9 + i] * xv[i];
    vf += wv[c * 9 + i] * xv[i];
  }
  const size_t qi = (size_t)(b * DIM + c) * HW + p;
  qh[qi] = (_Float16)qf;
  kh[qi] = (_Float16)kf;
  vT[((size_t)b * HW + p) * DIM + c] = (_Float16)vf;

  // block-wide sum of squares -> one atomic per block
  float sq = qf * qf, sk = kf * kf;
  #pragma unroll
  for (int off = 16; off >= 1; off >>= 1) {
    sq += __shfl_xor(sq, off, 32);
    sk += __shfl_xor(sk, off, 32);
  }
  __shared__ float lq[8], lk[8];
  const int wid = tid >> 5, lane = tid & 31;
  if (lane == 0) { lq[wid] = sq; lk[wid] = sk; }
  __syncthreads();
  if (tid == 0) {
    float aq = 0.f, ak = 0.f;
    #pragma unroll
    for (int i = 0; i < 8; ++i) { aq += lq[i]; ak += lk[i]; }
    atomicAdd(&ssq[b * DIM + c], aq);
    atomicAdd(&ssk[b * DIM + c], ak);
  }
}

// -------- Kernel 2: attn = q . k^T  (32x32 per bh, K = 16384, split) --------
// grid BH*NSLICE blocks of 32 (one wave, EXEC all-1s for WMMA).
__global__ void k_attn_gemm(const _Float16* __restrict__ qh,
                            const _Float16* __restrict__ kh,
                            float* __restrict__ attn) {
  const int lane  = threadIdx.x & 31;
  const int bh    = blockIdx.x / NSLICE;
  const int slice = blockIdx.x % NSLICE;
  const int k0    = slice * (HW / NSLICE);
  const _Float16* qb = qh + (size_t)bh * CH * HW;
  const _Float16* kb = kh + (size_t)bh * CH * HW;
  const int m = lane & 15;
  v8f c00 = {}, c01 = {}, c10 = {}, c11 = {};
  for (int k = 0; k < HW / NSLICE; k += 32) {
    const int kbs = k0 + k;
    const v16h a0 = load_frag16(qb + (size_t)m * HW,        kbs, lane);
    const v16h a1 = load_frag16(qb + (size_t)(m + 16) * HW, kbs, lane);
    const v16h b0 = load_frag16(kb + (size_t)m * HW,        kbs, lane);
    const v16h b1 = load_frag16(kb + (size_t)(m + 16) * HW, kbs, lane);
    c00 = __builtin_amdgcn_wmma_f32_16x16x32_f16(false, a0, false, b0, (short)0, c00, false, false);
    c01 = __builtin_amdgcn_wmma_f32_16x16x32_f16(false, a0, false, b1, (short)0, c01, false, false);
    c10 = __builtin_amdgcn_wmma_f32_16x16x32_f16(false, a1, false, b0, (short)0, c10, false, false);
    c11 = __builtin_amdgcn_wmma_f32_16x16x32_f16(false, a1, false, b1, (short)0, c11, false, false);
  }
  float* ab = attn + (size_t)bh * CH * CH;
  const int mb = (lane >> 4) * 8;   // D: lane holds N=lane%16, M=mb+v
  const int n  = lane & 15;
  #pragma unroll
  for (int v = 0; v < 8; ++v) {
    atomicAdd(&ab[(mb + v) * CH + n],           c00[v]);
    atomicAdd(&ab[(mb + v) * CH + 16 + n],      c01[v]);
    atomicAdd(&ab[(16 + mb + v) * CH + n],      c10[v]);
    atomicAdd(&ab[(16 + mb + v) * CH + 16 + n], c11[v]);
  }
}

// -------- Kernel 3: fold 1/(||q|| ||k||) + softmax over d, emit TRANSPOSED f16 --------
// smT[bh][d][c] = softmax(attn)[c][d]; grid BH*CH blocks of 32.
__global__ void k_softmax(const float* __restrict__ attn,
                          const float* __restrict__ ssq, const float* __restrict__ ssk,
                          _Float16* __restrict__ smT16) {
  const int d  = threadIdx.x;        // 0..31
  const int c  = blockIdx.x & 31;
  const int bh = blockIdx.x >> 5;
  const float rq = 1.f / fmaxf(sqrtf(ssq[bh * CH + c]), 1e-12f);
  const float rk = 1.f / fmaxf(sqrtf(ssk[bh * CH + d]), 1e-12f);
  const float v  = attn[((size_t)bh * CH + c) * CH + d] * rq * rk;
  float mx = v;
  #pragma unroll
  for (int off = 16; off >= 1; off >>= 1) mx = fmaxf(mx, __shfl_xor(mx, off, 32));
  const float e = __expf(v - mx);
  float s = e;
  #pragma unroll
  for (int off = 16; off >= 1; off >>= 1) s += __shfl_xor(s, off, 32);
  smT16[((size_t)bh * CH + d) * CH + c] = (_Float16)(e / s);
}

// -------- Kernel 4: W_eff[b][:, h*32:+32] = wp[:, h*32:+32] (192x32) x sm (32x32) --------
// Folds the 1x1 projection into the per-head attn matrix (out = (wp.blockdiag(sm)).v).
// grid BH*3 blocks of 32: og in 0..2 covers 4 o-tiles each; 2 d-tiles; K=32 (1 step).
__global__ void k_weff(const _Float16* __restrict__ wp16,
                       const _Float16* __restrict__ smT16,
                       _Float16* __restrict__ weff16) {
  const int lane = threadIdx.x & 31;
  const int og = blockIdx.x % 3;
  const int bh = blockIdx.x / 3;
  const int b = bh / HEADS, h = bh % HEADS;
  const _Float16* smT = smT16 + (size_t)bh * CH * CH;
  const int m = lane & 15;
  // B fragments: columns d of sm -> rows of smT
  const v16h b0 = load_frag16(smT + (size_t)m * CH, 0, lane);
  const v16h b1 = load_frag16(smT + (size_t)(m + 16) * CH, 0, lane);
  v8f acc[4][2] = {{{}, {}}, {{}, {}}, {{}, {}}, {{}, {}}};
  #pragma unroll
  for (int ot = 0; ot < 4; ++ot) {
    const int o = og * 64 + ot * 16 + m;
    const v16h af = load_frag16(wp16 + (size_t)o * DIM, h * CH, lane);
    acc[ot][0] = __builtin_amdgcn_wmma_f32_16x16x32_f16(false, af, false, b0, (short)0, acc[ot][0], false, false);
    acc[ot][1] = __builtin_amdgcn_wmma_f32_16x16x32_f16(false, af, false, b1, (short)0, acc[ot][1], false, false);
  }
  _Float16* wb = weff16 + (size_t)b * DIM * DIM;
  const int mb = (lane >> 4) * 8;
  const int n  = lane & 15;
  #pragma unroll
  for (int nt = 0; nt < 2; ++nt) {
    const int cglob = h * CH + nt * 16 + n;
    #pragma unroll
    for (int ot = 0; ot < 4; ++ot) {
      const int ob = og * 64 + ot * 16 + mb;
      #pragma unroll
      for (int v = 0; v < 8; ++v)
        wb[(size_t)(ob + v) * DIM + cglob] = (_Float16)acc[ot][nt][v];
    }
  }
}

// -------- Kernel 5: out[b] = W_eff[b] (192x192) x v[b] (192x16384) + bp --------
// One wave per 16-wide n-tile holds all 12 o-tile accumulators. Final stores
// are non-temporal (write-once, never re-read) to avoid evicting vT from L2.
// grid 8192 blocks of 32 (global n-tiles across batch).
__global__ void k_proj_final(const _Float16* __restrict__ weff16,
                             const _Float16* __restrict__ vT,
                             const float* __restrict__ bp,
                             float* __restrict__ out) {
  const int lane = threadIdx.x & 31;
  const int nt = blockIdx.x;                 // global n-tile 0..8191
  const int b  = nt / (HW / 16);             // uniform across wave
  const int m  = lane & 15;
  const _Float16* brow = vT + (size_t)(nt * 16 + m) * DIM;   // row n of v^T
  const _Float16* wA   = weff16 + (size_t)b * DIM * DIM;
  v8f acc[12] = {{}, {}, {}, {}, {}, {}, {}, {}, {}, {}, {}, {}};
  #pragma unroll
  for (int kk = 0; kk < 6; ++kk) {
    const v16h bf = load_frag16(brow, kk * 32, lane);
    #pragma unroll
    for (int ot = 0; ot < 12; ++ot) {
      const v16h af = load_frag16(wA + (size_t)(ot * 16 + m) * DIM, kk * 32, lane);
      acc[ot] = __builtin_amdgcn_wmma_f32_16x16x32_f16(false, af, false, bf, (short)0, acc[ot], false, false);
    }
  }
  const int nl = (nt * 16 + (lane & 15)) & (HW - 1);
  const int mb = (lane >> 4) * 8;
  #pragma unroll
  for (int ot = 0; ot < 12; ++ot) {
    const int ob = ot * 16 + mb;
    #pragma unroll
    for (int v = 0; v < 8; ++v) {
      const int o = ob + v;
      __builtin_nontemporal_store(acc[ot][v] + bp[o],
                                  out + ((size_t)(b * DIM + o)) * HW + nl);
    }
  }
}

extern "C" void kernel_launch(void* const* d_in, const int* in_sizes, int n_in,
                              void* d_out, int out_size, void* d_ws, size_t ws_size,
                              hipStream_t stream) {
  const float* x  = (const float*)d_in[0];
  const float* wq = (const float*)d_in[1];
  const float* bq = (const float*)d_in[2];
  const float* wk = (const float*)d_in[3];
  const float* bk = (const float*)d_in[4];
  const float* wv = (const float*)d_in[5];
  const float* bv = (const float*)d_in[6];
  const float* wp = (const float*)d_in[7];
  const float* bp = (const float*)d_in[8];
  float* out = (float*)d_out;

  // workspace layout (bytes); total ~152 MB (fits global L2 = 192 MB)
  char* ws = (char*)d_ws;
  const size_t qkvB = (size_t)BATCH * DIM * HW * sizeof(_Float16);  // 50331648
  _Float16* qh = (_Float16*)(ws);
  _Float16* kh = (_Float16*)(ws + qkvB);
  _Float16* vT = (_Float16*)(ws + 2 * qkvB);
  char* p = ws + 3 * qkvB;
  float*    attn   = (float*)p;      p += (size_t)BH * CH * CH * sizeof(float);
  _Float16* smT16  = (_Float16*)p;   p += (size_t)BH * CH * CH * sizeof(_Float16);
  float*    ssq    = (float*)p;      p += (size_t)BATCH * DIM * sizeof(float);
  float*    ssk    = (float*)p;      p += (size_t)BATCH * DIM * sizeof(float);
  _Float16* wp16   = (_Float16*)p;   p += (size_t)DIM * DIM * sizeof(_Float16);
  _Float16* weff16 = (_Float16*)p;

  // 0: init accumulators + wp conversion
  k_init<<<(BH * CH * CH + 255) / 256, 256, 0, stream>>>(attn, ssq, ssk, wp, wp16);
  // 1: fused depthwise convs (q,k row-major f16; v channel-minor f16)
  k_dwconv_qkv<<<dim3(HW / 256, DIM, BATCH), 256, 0, stream>>>(
      x, wq, bq, wk, bk, wv, bv, qh, kh, vT, ssq, ssk);
  // 2: attn = q.k^T (sliced-K, atomic reduce)
  k_attn_gemm<<<BH * NSLICE, 32, 0, stream>>>(qh, kh, attn);
  // 3: normalization fold + softmax (emit transposed f16)
  k_softmax<<<BH * CH, 32, 0, stream>>>(attn, ssq, ssk, smT16);
  // 4: fold projection into attn: W_eff = wp . blockdiag(sm)
  k_weff<<<BH * 3, 32, 0, stream>>>(wp16, smT16, weff16);
  // 5: out = W_eff . v + bp  (single big GEMM, writes d_out directly)
  k_proj_final<<<8192, 32, 0, stream>>>(weff16, vT, bp, out);
}
